// HGRU_78305843740963
// MI455X (gfx1250) — matrix operand order repaired and checked by reading, not compiled, not run on hardware
//
#include <hip/hip_runtime.h>
#include <hip/hip_bf16.h>
#include <cstddef>

// ---------------- problem constants ----------------
#define BB   16384
#define TU   72
#define TY   24
#define IND  10
#define HD   64
#define NB   4
#define HB   16
#define NHL  4

#define WAVES 4                 // waves per block, each owns a 16-row batch tile
#define THREADS (WAVES * 32)
#define BLOCKS (BB / (WAVES * 16))

typedef __attribute__((ext_vector_type(16))) _Float16 v16h;
typedef __attribute__((ext_vector_type(8)))  _Float16 v8h;
typedef __attribute__((ext_vector_type(8)))  float    v8f;

struct KParams {
    const float* u;
    const float *in_Wih, *in_Whh, *in_bih, *in_bhh;
    const float *sh_Wih, *sh_Whh, *sh_bih, *sh_bhh;
    const float *br_Wih[NB][NHL], *br_Whh[NB][NHL];
    const float *br_bih[NB][NHL], *br_bhh[NB][NHL];
    const float *lin_W[NB], *lin_b[NB];
    float* out;
};

// ---------------- fast transcendentals (hardware trans unit) ----------------
__device__ __forceinline__ float sigmoid_fast(float x) {
    // v_exp_f32 + v_add + v_rcp_f32 (no IEEE divide expansion)
    return __builtin_amdgcn_rcpf(1.0f + __expf(-x));
}
__device__ __forceinline__ float tanh_fast(float x) {
#if __has_builtin(__builtin_amdgcn_tanhf)
    return __builtin_amdgcn_tanhf(x);          // V_TANH_F32
#else
    return 2.0f * sigmoid_fast(2.0f * x) - 1.0f;
#endif
}

// ---------------- fragment helpers ----------------
// A fragment (16x32 f16), documented layout:
// lane L: M = L&15, hk = L>>4; halves[0..7] = K hk*8..hk*8+7, halves[8..15] = K 16+hk*8..+7
__device__ __forceinline__ v16h load_a_full(const _Float16* buf, int ld, int kk, int lane) {
    const int m = lane & 15, hk = lane >> 4;
    const _Float16* p = buf + m * ld + kk * 32;
    v8h c0 = *(const v8h*)(p + hk * 8);
    v8h c1 = *(const v8h*)(p + 16 + hk * 8);
    return __builtin_shufflevector(c0, c1, 0,1,2,3,4,5,6,7,8,9,10,11,12,13,14,15);
}

// A fragment for a 16x16 tile zero-padded to K=32 (ld = 16)
__device__ __forceinline__ v16h load_a_half16(const _Float16* buf, int lane) {
    const int m = lane & 15, hk = lane >> 4;
    v8h c0 = *(const v8h*)(buf + m * 16 + hk * 8);
    v8h z  = {};
    return __builtin_shufflevector(c0, z, 0,1,2,3,4,5,6,7,8,9,10,11,12,13,14,15);
}

// B fragment from pre-swizzled LDS tile: tile = 512 halves, lane-major, 32 contiguous bytes/lane
__device__ __forceinline__ v16h load_b(const _Float16* w, int tile, int lane) {
    const _Float16* p = w + tile * 512 + lane * 16;
    v8h c0 = *(const v8h*)(p);
    v8h c1 = *(const v8h*)(p + 8);
    return __builtin_shufflevector(c0, c1, 0,1,2,3,4,5,6,7,8,9,10,11,12,13,14,15);
}

// A fragment for the raw input u (K = 10, padded to 32)
__device__ __forceinline__ v16h load_a_u(const float* u, int rowBase, int t, int lane) {
    const int m = lane & 15, hk = lane >> 4;
    const float* up = u + ((size_t)(rowBase + m) * TU + t) * IND;
    v16h a;
#pragma unroll
    for (int i = 0; i < 8; ++i) {
        const int k = hk * 8 + i;
        float v = up[k < IND ? k : 0];          // clamped address: never OOB
        a[i] = (_Float16)(k < IND ? v : 0.0f);  // masked value
    }
#pragma unroll
    for (int i = 8; i < 16; ++i) a[i] = (_Float16)0.0f;
    return a;
}

// Swizzle a row-major weight W[gateDim][Din] into B-fragment tiles (K padded to nK*32).
// B[k][n] = W[ntile*16+n][ktile*32+k]; lane l2: N=l2&15, K = (l2>>4)*8 + i (i<8), 16 + (l2>>4)*8 + i-8 (i>=8).
__device__ __forceinline__ void swz(const float* W, _Float16* dst,
                                    int gateDim, int Din, int nK, int tid, int nth) {
    const int total = (gateDim / 16) * nK * 512;
    for (int idx = tid; idx < total; idx += nth) {
        const int tile = idx >> 9;
        const int slot = idx & 511;
        const int l2 = slot >> 4;
        const int i  = slot & 15;
        const int nt = tile / nK;
        const int kk = tile - nt * nK;
        const int n   = l2 & 15;
        const int hkk = l2 >> 4;
        const int gk  = kk * 32 + (i < 8 ? hkk * 8 + i : 16 + hkk * 8 + (i - 8));
        const float v = (gk < Din) ? W[(nt * 16 + n) * Din + gk] : 0.0f;
        dst[idx] = (_Float16)v;
    }
}

// One GRU timestep for a 16-row tile. KX = # K-chunks of x fragments, DH = hidden size.
// Accumulators start from inline-0 C; biases are applied in the gate epilogue:
//   brz[0..DH)   = bih_r + bhh_r,  brz[DH..2DH) = bih_z + bhh_z
//   bnx[0..DH)   = bih_n,          bnh[0..DH)   = bhh_n
template <int KX, int DH>
__device__ __forceinline__ void gru_step(const v16h* xa, _Float16* hbuf,
                                         const _Float16* wih, const _Float16* whh,
                                         const float* brz, const float* bnx,
                                         const float* bnh, int lane) {
    constexpr int KH = (DH + 31) / 32;
    constexpr int TJ = DH / 16;
    v16h ha[KH];
    if constexpr (DH == 64) {
        ha[0] = load_a_full(hbuf, 64, 0, lane);
        ha[1] = load_a_full(hbuf, 64, 1, lane);
    } else {
        ha[0] = load_a_half16(hbuf, lane);
    }
    const int n = lane & 15;
    const int rbase = (lane >> 4) * 8;
#pragma unroll
    for (int j = 0; j < TJ; ++j) {
        v8f cx[3], ch[3];
#pragma unroll
        for (int g = 0; g < 3; ++g) {               // gate order: r, z, n
            const int nt = g * TJ + j;
            v8f ax = {};                            // inline-0 C operand
#pragma unroll
            for (int kk = 0; kk < KX; ++kk)
                ax = __builtin_amdgcn_wmma_f32_16x16x32_f16(
                    false, xa[kk], false, load_b(wih, nt * KX + kk, lane),
                    (short)0, ax, false, false);
            cx[g] = ax;
            v8f ah = {};
#pragma unroll
            for (int kk = 0; kk < KH; ++kk)
                ah = __builtin_amdgcn_wmma_f32_16x16x32_f16(
                    false, ha[kk], false, load_b(whh, nt * KH + kk, lane),
                    (short)0, ah, false, false);
            ch[g] = ah;
        }
        const float b_r = brz[j * 16 + n];
        const float b_z = brz[DH + j * 16 + n];
        const float b_x = bnx[j * 16 + n];
        const float b_h = bnh[j * 16 + n];
#pragma unroll
        for (int r = 0; r < 8; ++r) {
            const float rr = sigmoid_fast(cx[0][r] + ch[0][r] + b_r);
            const float zz = sigmoid_fast(cx[1][r] + ch[1][r] + b_z);
            const float nn = tanh_fast(cx[2][r] + b_x + rr * (ch[2][r] + b_h));
            const int m = rbase + r;
            const int idx = m * DH + j * 16 + n;
            const float hold = (float)hbuf[idx];
            hbuf[idx] = (_Float16)(nn + zz * (hold - nn));   // (1-z)*n + z*h
        }
    }
}

// ---------------- main fused kernel ----------------
__global__ __launch_bounds__(THREADS) void hgru_fused(KParams p) {
    // Weights, pre-swizzled into B-fragment tiles (f16), shared by all waves in block.
    __shared__ __attribute__((aligned(16))) _Float16 wIN_ih[12 * 512];     // 192x10  (K pad 32)
    __shared__ __attribute__((aligned(16))) _Float16 wIN_hh[24 * 512];     // 192x64
    __shared__ __attribute__((aligned(16))) _Float16 wSH_ih[24 * 512];     // 192x64
    __shared__ __attribute__((aligned(16))) _Float16 wSH_hh[24 * 512];     // 192x64
    __shared__ __attribute__((aligned(16))) _Float16 wBR[NB * 27 * 512];   // per branch 27 tiles
    // Bias arrays: [rz combined (2*DH)], [n-gate x part (DH)], [n-gate h part (DH)]
    __shared__ float bIN_rz[128], bIN_nx[64], bIN_nh[64];
    __shared__ float bSH_rz[128], bSH_nx[64], bSH_nh[64];
    __shared__ float bBR_rz[NB][NHL][32], bBR_nx[NB][NHL][16], bBR_nh[NB][NHL][16];
    __shared__ float sLinW[NB][16];
    __shared__ float sLinB[NB];
    // Recurrent states (per wave), f16 row-major
    __shared__ __attribute__((aligned(16))) _Float16 hIN[WAVES * 16 * 64];
    __shared__ __attribute__((aligned(16))) _Float16 hSH[WAVES * 16 * 64];
    __shared__ __attribute__((aligned(16))) _Float16 hBR[WAVES * NB * NHL * 16 * 16];

    const int tid = threadIdx.x;
    const int nth = THREADS;
    const int wave = tid >> 5;
    const int lane = tid & 31;
    const int rowBase = blockIdx.x * (WAVES * 16) + wave * 16;

    // ---- stage weights / biases into LDS, zero states ----
    swz(p.in_Wih, wIN_ih, 192, IND, 1, tid, nth);
    swz(p.in_Whh, wIN_hh, 192, HD, 2, tid, nth);
    swz(p.sh_Wih, wSH_ih, 192, HD, 2, tid, nth);
    swz(p.sh_Whh, wSH_hh, 192, HD, 2, tid, nth);
#pragma unroll
    for (int b = 0; b < NB; ++b) {
        _Float16* base = wBR + b * (27 * 512);
        swz(p.br_Wih[b][0], base, 48, HD, 2, tid, nth);                    // tiles 0..5
        swz(p.br_Whh[b][0], base + 6 * 512, 48, HB, 1, tid, nth);          // tiles 6..8
#pragma unroll
        for (int l = 1; l < NHL; ++l) {
            swz(p.br_Wih[b][l], base + (9 + (l - 1) * 6) * 512, 48, HB, 1, tid, nth);
            swz(p.br_Whh[b][l], base + (12 + (l - 1) * 6) * 512, 48, HB, 1, tid, nth);
        }
    }
    for (int i = tid; i < 64; i += nth) {
        bIN_rz[i]      = p.in_bih[i]       + p.in_bhh[i];        // r
        bIN_rz[64 + i] = p.in_bih[64 + i]  + p.in_bhh[64 + i];   // z
        bIN_nx[i]      = p.in_bih[128 + i];
        bIN_nh[i]      = p.in_bhh[128 + i];
        bSH_rz[i]      = p.sh_bih[i]       + p.sh_bhh[i];
        bSH_rz[64 + i] = p.sh_bih[64 + i]  + p.sh_bhh[64 + i];
        bSH_nx[i]      = p.sh_bih[128 + i];
        bSH_nh[i]      = p.sh_bhh[128 + i];
    }
    for (int i = tid; i < 16; i += nth) {
        for (int b = 0; b < NB; ++b) {
            for (int l = 0; l < NHL; ++l) {
                bBR_rz[b][l][i]      = p.br_bih[b][l][i]      + p.br_bhh[b][l][i];
                bBR_rz[b][l][16 + i] = p.br_bih[b][l][16 + i] + p.br_bhh[b][l][16 + i];
                bBR_nx[b][l][i]      = p.br_bih[b][l][32 + i];
                bBR_nh[b][l][i]      = p.br_bhh[b][l][32 + i];
            }
            sLinW[b][i] = p.lin_W[b][i];
        }
    }
    if (tid < NB) sLinB[tid] = p.lin_b[tid][0];
    for (int i = tid; i < WAVES * 16 * 64; i += nth) { hIN[i] = (_Float16)0.0f; hSH[i] = (_Float16)0.0f; }
    for (int i = tid; i < WAVES * NB * NHL * 16 * 16; i += nth) hBR[i] = (_Float16)0.0f;
    __syncthreads();

    _Float16* myHIN = hIN + wave * (16 * 64);
    _Float16* myHSH = hSH + wave * (16 * 64);

    // ---- fused time loop: input GRU -> shared GRU -> (branches for last TY steps) ----
    for (int t = 0; t < TU; ++t) {
        // input layer (Din=10 padded to 32)
        v16h xa0[1];
        xa0[0] = load_a_u(p.u, rowBase, t, lane);
        gru_step<1, 64>(xa0, myHIN, wIN_ih, wIN_hh, bIN_rz, bIN_nx, bIN_nh, lane);

        // shared layer (Din=64)
        v16h xs[2];
        xs[0] = load_a_full(myHIN, 64, 0, lane);
        xs[1] = load_a_full(myHIN, 64, 1, lane);
        gru_step<2, 64>(xs, myHSH, wSH_ih, wSH_hh, bSH_rz, bSH_nx, bSH_nh, lane);

        if (t >= TU - TY) {
            v16h xb[2];
            xb[0] = load_a_full(myHSH, 64, 0, lane);
            xb[1] = load_a_full(myHSH, 64, 1, lane);
#pragma unroll
            for (int b = 0; b < NB; ++b) {
                const _Float16* wb = wBR + b * (27 * 512);
                _Float16* hb0 = hBR + ((wave * NB + b) * NHL + 0) * 256;
                gru_step<2, 16>(xb, hb0, wb, wb + 6 * 512,
                                bBR_rz[b][0], bBR_nx[b][0], bBR_nh[b][0], lane);
#pragma unroll
                for (int l = 1; l < NHL; ++l) {
                    v16h xl[1];
                    xl[0] = load_a_half16(hBR + ((wave * NB + b) * NHL + (l - 1)) * 256, lane);
                    gru_step<1, 16>(xl, hBR + ((wave * NB + b) * NHL + l) * 256,
                                    wb + (9 + (l - 1) * 6) * 512,
                                    wb + (12 + (l - 1) * 6) * 512,
                                    bBR_rz[b][l], bBR_nx[b][l], bBR_nh[b][l], lane);
                }
                // linear head: 16 -> 1
                const _Float16* h4 = hBR + ((wave * NB + b) * NHL + (NHL - 1)) * 256;
                const int m = lane & 15;
                float acc = sLinB[b];
#pragma unroll
                for (int c = 0; c < 16; ++c) acc += (float)h4[m * 16 + c] * sLinW[b][c];
                if (lane < 16)
                    p.out[(size_t)b * BB * TY + (size_t)(rowBase + m) * TY + (t - (TU - TY))] = acc;
            }
        }
    }
}

// ---------------- host launcher ----------------
extern "C" void kernel_launch(void* const* d_in, const int* in_sizes, int n_in,
                              void* d_out, int out_size, void* d_ws, size_t ws_size,
                              hipStream_t stream) {
    (void)n_in; (void)out_size; (void)d_ws; (void)ws_size;
    KParams kp;
    auto F = [&](int i) { return (const float*)d_in[i]; };
    const int U_SIZE = BB * TU * IND;
    if (in_sizes[0] == U_SIZE) {
        // insertion order: u; input{Wih,Whh,bih,bhh}; shared{...};
        // branches[b]{grus[l]{Wih,Whh,bih,bhh}, lin{W,b}}
        int i = 0;
        kp.u = F(i++);
        kp.in_Wih = F(i++); kp.in_Whh = F(i++); kp.in_bih = F(i++); kp.in_bhh = F(i++);
        kp.sh_Wih = F(i++); kp.sh_Whh = F(i++); kp.sh_bih = F(i++); kp.sh_bhh = F(i++);
        for (int b = 0; b < NB; ++b) {
            for (int l = 0; l < NHL; ++l) {
                kp.br_Wih[b][l] = F(i++); kp.br_Whh[b][l] = F(i++);
                kp.br_bih[b][l] = F(i++); kp.br_bhh[b][l] = F(i++);
            }
            kp.lin_W[b] = F(i++); kp.lin_b[b] = F(i++);
        }
    } else {
        // jax tree-sorted order: branches (Whh,Wih,bhh,bih per layer; lin W,b),
        // then input (Whh,Wih,bhh,bih), shared, then u.
        int i = 0;
        for (int b = 0; b < NB; ++b) {
            for (int l = 0; l < NHL; ++l) {
                kp.br_Whh[b][l] = F(i++); kp.br_Wih[b][l] = F(i++);
                kp.br_bhh[b][l] = F(i++); kp.br_bih[b][l] = F(i++);
            }
            kp.lin_W[b] = F(i++); kp.lin_b[b] = F(i++);
        }
        kp.in_Whh = F(i++); kp.in_Wih = F(i++); kp.in_bhh = F(i++); kp.in_bih = F(i++);
        kp.sh_Whh = F(i++); kp.sh_Wih = F(i++); kp.sh_bhh = F(i++); kp.sh_bih = F(i++);
        kp.u = F(i++);
    }
    kp.out = (float*)d_out;
    hgru_fused<<<BLOCKS, THREADS, 0, stream>>>(kp);
}
